// BlockPrototypeMemory_67164698574954
// MI455X (gfx1250) — matrix operand order repaired and checked by reading, not compiled, not run on hardware
//
#include <hip/hip_runtime.h>
#include <hip/hip_bf16.h>
#include <stdint.h>

// ---- problem dims ----
#define P_PROTO 512
#define NBLK    16
#define DB      128
#define HID     512            // 4*DB
#define BN      32768          // B*N = 8*4096
#define D_MODEL 2048
#define LN_EPS  1e-5f
#define QSCALE  0.08838834764831845f   // DB^-0.5

typedef __attribute__((ext_vector_type(16))) __bf16        v16bf;
typedef __attribute__((ext_vector_type(8)))  float         v8f;
typedef __attribute__((ext_vector_type(8)))  unsigned int  v8u;
typedef __attribute__((ext_vector_type(4)))  unsigned int  v4u;
typedef __attribute__((ext_vector_type(8)))  int           v8i_t;
typedef __attribute__((ext_vector_type(4)))  int           v4i_t;

__device__ __forceinline__ unsigned short f2bf(float f) {
    union { float f; uint32_t u; } v; v.f = f;
    uint32_t u = v.u;
    u += 0x7FFFu + ((u >> 16) & 1u);     // round-to-nearest-even
    return (unsigned short)(u >> 16);
}

__device__ __forceinline__ v16bf frag_from(uint4 a, uint4 b) {
    v8u w;
    w[0] = a.x; w[1] = a.y; w[2] = a.z; w[3] = a.w;
    w[4] = b.x; w[5] = b.y; w[6] = b.z; w[7] = b.w;
    return __builtin_bit_cast(v16bf, w);
}

// ============================================================================
// Kernel 1: mem MLP (128->512->512->512->128, relu) + block LN per row.
// One workgroup (128 threads) per (p,h) row. Writes bf16 K [h][p][d] and
// KT [h][d][p] so attention B-fragments are contiguous for both matmuls.
// ============================================================================
__global__ void mem_mlp_ln_kernel(const float* __restrict__ mem_params,
                                  const float* __restrict__ w1, const float* __restrict__ b1,
                                  const float* __restrict__ w2, const float* __restrict__ b2,
                                  const float* __restrict__ w3, const float* __restrict__ b3,
                                  const float* __restrict__ w4, const float* __restrict__ b4,
                                  unsigned short* __restrict__ k_bf,
                                  unsigned short* __restrict__ kt_bf) {
    __shared__ float xs[DB], h1[HID], h2[HID], h3[HID], os[DB], red[DB];
    const int t   = threadIdx.x;          // 0..127
    const int row = blockIdx.x;           // p*16 + h  (mem_params is [P][NB][DB])
    const int p   = row >> 4;
    const int h   = row & 15;

    xs[t] = mem_params[(size_t)row * DB + t];
    __syncthreads();

    #pragma unroll
    for (int jj = 0; jj < 4; ++jj) {
        const int j = t + jj * 128;
        const float* wr = w1 + (size_t)j * DB;
        float acc = b1[j];
        #pragma unroll 8
        for (int k = 0; k < DB; ++k) acc += wr[k] * xs[k];
        h1[j] = fmaxf(acc, 0.f);
    }
    __syncthreads();

    #pragma unroll
    for (int jj = 0; jj < 4; ++jj) {
        const int j = t + jj * 128;
        const float* wr = w2 + (size_t)j * HID;
        float acc = b2[j];
        #pragma unroll 8
        for (int k = 0; k < HID; ++k) acc += wr[k] * h1[k];
        h2[j] = fmaxf(acc, 0.f);
    }
    __syncthreads();

    #pragma unroll
    for (int jj = 0; jj < 4; ++jj) {
        const int j = t + jj * 128;
        const float* wr = w3 + (size_t)j * HID;
        float acc = b3[j];
        #pragma unroll 8
        for (int k = 0; k < HID; ++k) acc += wr[k] * h2[k];
        h3[j] = fmaxf(acc, 0.f);
    }
    __syncthreads();

    {
        const float* wr = w4 + (size_t)t * HID;
        float acc = b4[t];
        #pragma unroll 8
        for (int k = 0; k < HID; ++k) acc += wr[k] * h3[k];
        os[t] = acc;
    }
    __syncthreads();

    // block LayerNorm over the 128 outputs
    red[t] = os[t];
    __syncthreads();
    for (int s = 64; s > 0; s >>= 1) { if (t < s) red[t] += red[t + s]; __syncthreads(); }
    const float mu = red[0] * (1.f / DB);
    __syncthreads();
    const float e = os[t] - mu;
    red[t] = e * e;
    __syncthreads();
    for (int s = 64; s > 0; s >>= 1) { if (t < s) red[t] += red[t + s]; __syncthreads(); }
    const float var = red[0] * (1.f / DB);
    const float y = e * rsqrtf(var + LN_EPS);

    const unsigned short yb = f2bf(y);
    k_bf [((size_t)h * P_PROTO + p) * DB + t]      = yb;   // [h][p][d]
    kt_bf[((size_t)h * DB + t) * P_PROTO + p]      = yb;   // [h][d][p]
}

// ============================================================================
// Kernel 2: query block-LN + DB^-0.5 scale -> bf16 [h][token][128].
// One wave32 per (h, token) row of 128; 4 floats per lane.
// ============================================================================
__global__ void q_ln_kernel(const float* __restrict__ q, unsigned short* __restrict__ q_bf) {
    const int lane = threadIdx.x & 31;
    const int wave = threadIdx.x >> 5;
    const long r = (long)blockIdx.x * 8 + wave;   // 0 .. 524287
    const int h = (int)(r >> 15);
    const int token = (int)(r & (BN - 1));

    const float4 v = ((const float4*)(q + (size_t)token * D_MODEL + h * DB))[lane];
    float s = v.x + v.y + v.z + v.w;
    #pragma unroll
    for (int m = 1; m < 32; m <<= 1) s += __shfl_xor(s, m, 32);
    const float mu = s * (1.f / DB);
    const float ex = v.x - mu, ey = v.y - mu, ez = v.z - mu, ew = v.w - mu;
    float ss = ex * ex + ey * ey + ez * ez + ew * ew;
    #pragma unroll
    for (int m = 1; m < 32; m <<= 1) ss += __shfl_xor(ss, m, 32);
    const float rs = rsqrtf(ss * (1.f / DB) + LN_EPS) * QSCALE;

    const uint32_t w0 = (uint32_t)f2bf(ex * rs) | ((uint32_t)f2bf(ey * rs) << 16);
    const uint32_t w1 = (uint32_t)f2bf(ez * rs) | ((uint32_t)f2bf(ew * rs) << 16);
    *(uint2*)(q_bf + (size_t)r * DB + lane * 4) = make_uint2(w0, w1);
}

// ============================================================================
// Kernel 3: fused flash-style block attention.
// grid = (BN/128, NBLK); block = 256 (8 waves). Wave w owns tokens
// tileBase + 16w .. +15. TDM DMAs K [512][128] and KT [128][512] (bf16) into
// LDS (256 KB of the 320 KB WGP LDS), then 16 iterations of 32 prototypes:
// S via 8x v_wmma_f32_16x16x32_bf16, online softmax, P via LDS bounce, O via
// 8 more WMMAs.
// ============================================================================
__global__ __launch_bounds__(256, 1)
void block_attn_kernel(const unsigned short* __restrict__ q_bf,
                       const unsigned short* __restrict__ k_bf,
                       const unsigned short* __restrict__ kt_bf,
                       float* __restrict__ out) {
    extern __shared__ char smem[];
    // layout: [0,131072) K [512][128] bf16 ; [131072,262144) KT [128][512] bf16
    //         [262144,270336) per-wave P bounce [8][16][32] bf16
    unsigned short* Kl  = (unsigned short*)smem;
    unsigned short* KTl = (unsigned short*)(smem + 131072);

    const int h        = blockIdx.y;
    const int tileBase = blockIdx.x * 128;
    const int tid  = threadIdx.x;
    const int wave = tid >> 5;
    const int lane = tid & 31;

    unsigned short* Pb = (unsigned short*)(smem + 262144) + wave * 16 * 32;

    // ---- TDM: DMA K and KT tiles for this block into LDS (wave 0 only) ----
    if (wave == 0) {
        const uint64_t kaddr  = (uint64_t)(uintptr_t)(k_bf  + (size_t)h * P_PROTO * DB);
        const uint64_t ktaddr = (uint64_t)(uintptr_t)(kt_bf + (size_t)h * DB * P_PROTO);
        const v4i_t z4 = {0, 0, 0, 0};
        const v8i_t z8 = {0, 0, 0, 0, 0, 0, 0, 0};

        // D# group0: count=1 | lds_addr | global_addr | type=2
        v4u g0k = { 1u, 0u, (uint32_t)kaddr,
                    ((uint32_t)((kaddr >> 32) & 0x01FFFFFFu)) | 0x80000000u };
        v4u g0t = { 1u, 131072u, (uint32_t)ktaddr,
                    ((uint32_t)((ktaddr >> 32) & 0x01FFFFFFu)) | 0x80000000u };

        // D# group1: data_size=2B; K: dim0=128 stride0=128 dim1=512, tile 128x512
        v8i_t g1k = { (int)(1u << 16),
                      (int)(128u << 16),                 // tensor_dim0 lo16 @ [31:16]
                      (int)(512u << 16),                 // dim0 hi16=0 | tensor_dim1 lo16
                      (int)(128u << 16),                 // dim1 hi16=0 | tile_dim0
                      (int)(512u),                       // tile_dim1 | tile_dim2=0
                      (int)(128u),                       // tensor_dim0_stride lo32
                      0, 0 };
        // KT: dim0=512 stride0=512 dim1=128, tile 512x128
        v8i_t g1t = { (int)(1u << 16),
                      (int)(512u << 16),
                      (int)(128u << 16),
                      (int)(512u << 16),
                      (int)(128u),
                      (int)(512u),
                      0, 0 };

        __builtin_amdgcn_tensor_load_to_lds(g0k, g1k, z4, z4, z8, 0);
        __builtin_amdgcn_tensor_load_to_lds(g0t, g1t, z4, z4, z8, 0);
        __builtin_amdgcn_s_wait_tensorcnt(0);
    }
    __syncthreads();

    // ---- load persistent Q A-fragments (16 rows x 128, 4 chunks of K=32) ----
    const int rowA = tileBase + wave * 16 + (lane & 15);  // this lane's A row (token)
    const int kb   = (lane >> 4) * 16;                    // A-layout K run base
    const unsigned short* qrow = q_bf + ((size_t)h * BN + rowA) * DB;
    v16bf qa[4];
    #pragma unroll
    for (int c = 0; c < 4; ++c) {
        const uint4 a = *(const uint4*)(qrow + c * 32 + kb);
        const uint4 b = *(const uint4*)(qrow + c * 32 + kb + 8);
        qa[c] = frag_from(a, b);
    }

    v8f o_acc[8] = {};
    float m_run[8], l_run[8];
    #pragma unroll
    for (int i = 0; i < 8; ++i) { m_run[i] = -__builtin_inff(); l_run[i] = 0.f; }

    for (int j = 0; j < 16; ++j) {          // 32 prototypes per iteration
        const int pbase = j * 32;

        // ---- S = Q @ K^T : two 16x16 C tiles ----
        v8f s0 = {}, s1 = {};
        const int prow   = pbase + (lane & 15);
        #pragma unroll
        for (int c = 0; c < 4; ++c) {
            const int dstart = c * 32 + (lane >> 4) * 16;
            const uint4* bp0 = (const uint4*)(Kl + (size_t)prow * DB + dstart);
            const v16bf bf0 = frag_from(bp0[0], bp0[1]);
            s0 = __builtin_amdgcn_wmma_f32_16x16x32_bf16(false, qa[c], false, bf0,
                                                         (short)0, s0, false, false);
            const uint4* bp1 = (const uint4*)(Kl + (size_t)(prow + 16) * DB + dstart);
            const v16bf bf1 = frag_from(bp1[0], bp1[1]);
            s1 = __builtin_amdgcn_wmma_f32_16x16x32_bf16(false, qa[c], false, bf1,
                                                         (short)0, s1, false, false);
        }

        // ---- online softmax (row stats live in matching C VGPR index) ----
        float corr[8];
        #pragma unroll
        for (int i = 0; i < 8; ++i) {
            float t = fmaxf(s0[i], s1[i]);
            #pragma unroll
            for (int m = 1; m < 16; m <<= 1) t = fmaxf(t, __shfl_xor(t, m, 32));
            const float mn = fmaxf(m_run[i], t);
            corr[i] = __expf(m_run[i] - mn);
            m_run[i] = mn;
            const float p0 = __expf(s0[i] - mn);
            const float p1 = __expf(s1[i] - mn);
            s0[i] = p0; s1[i] = p1;
            float rsum = p0 + p1;
            #pragma unroll
            for (int m = 1; m < 16; m <<= 1) rsum += __shfl_xor(rsum, m, 32);
            l_run[i] = l_run[i] * corr[i] + rsum;
        }
        #pragma unroll
        for (int nt = 0; nt < 8; ++nt)
            #pragma unroll
            for (int i = 0; i < 8; ++i) o_acc[nt][i] *= corr[i];

        // ---- C-layout -> A-layout via per-wave LDS bounce (in-order DS pipe) ----
        {
            const int r0 = (lane >> 4) * 8;
            const int cc = lane & 15;
            #pragma unroll
            for (int i = 0; i < 8; ++i) {
                Pb[(r0 + i) * 32 + cc]      = f2bf(s0[i]);
                Pb[(r0 + i) * 32 + cc + 16] = f2bf(s1[i]);
            }
        }
        v16bf pa;
        {
            const int row = lane & 15;
            const uint4 a = *(const uint4*)(Pb + row * 32 + kb);
            const uint4 b = *(const uint4*)(Pb + row * 32 + kb + 8);
            pa = frag_from(a, b);
        }

        // ---- O += P(16x32) @ V(32x128), V rows = prototypes (from KT LDS) ----
        const int ps = pbase + (lane >> 4) * 16;
        #pragma unroll
        for (int nt = 0; nt < 8; ++nt) {
            const int dcol = nt * 16 + (lane & 15);
            const uint4* bp = (const uint4*)(KTl + (size_t)dcol * P_PROTO + ps);
            const v16bf vb = frag_from(bp[0], bp[1]);
            o_acc[nt] = __builtin_amdgcn_wmma_f32_16x16x32_bf16(false, pa, false, vb,
                                                                (short)0, o_acc[nt], false, false);
        }
    }

    // ---- epilogue: normalize by row sum, store f32 ----
    #pragma unroll
    for (int i = 0; i < 8; ++i) {
        const float inv = 1.f / l_run[i];
        const int token = tileBase + wave * 16 + i + 8 * (lane >> 4);
        #pragma unroll
        for (int nt = 0; nt < 8; ++nt) {
            out[(size_t)token * D_MODEL + h * DB + nt * 16 + (lane & 15)] = o_acc[nt][i] * inv;
        }
    }
}

// ============================================================================
extern "C" void kernel_launch(void* const* d_in, const int* in_sizes, int n_in,
                              void* d_out, int out_size, void* d_ws, size_t ws_size,
                              hipStream_t stream) {
    const float* q  = (const float*)d_in[0];
    const float* mp = (const float*)d_in[1];
    const float* w1 = (const float*)d_in[2];
    const float* b1 = (const float*)d_in[3];
    const float* w2 = (const float*)d_in[4];
    const float* b2 = (const float*)d_in[5];
    const float* w3 = (const float*)d_in[6];
    const float* b3 = (const float*)d_in[7];
    const float* w4 = (const float*)d_in[8];
    const float* b4 = (const float*)d_in[9];
    float* out = (float*)d_out;

    // workspace: K bf16 (2MB) | KT bf16 (2MB) | Q bf16 (128MB)
    unsigned short* k_bf  = (unsigned short*)d_ws;
    unsigned short* kt_bf = (unsigned short*)((char*)d_ws + 2097152);
    unsigned short* q_bf  = (unsigned short*)((char*)d_ws + 4194304);

    mem_mlp_ln_kernel<<<P_PROTO * NBLK, 128, 0, stream>>>(mp, w1, b1, w2, b2, w3, b3, w4, b4,
                                                          k_bf, kt_bf);
    q_ln_kernel<<<(NBLK * BN) / 8, 256, 0, stream>>>(q, q_bf);

    dim3 grid(BN / 128, NBLK);
    block_attn_kernel<<<grid, 256, 270336, stream>>>(q_bf, k_bf, kt_bf, out);
}
